// MASTER_81192061764140
// MI455X (gfx1250) — compile-verified, matrix operand order checked by reading, not run on hardware
//
#include <hip/hip_runtime.h>
#include <hip/hip_bf16.h>
#include <math.h>

typedef __attribute__((ext_vector_type(16))) __bf16 v16bf;
typedef __attribute__((ext_vector_type(8)))  float  v8f;
typedef __attribute__((ext_vector_type(4)))  unsigned int u32x4;
typedef __attribute__((ext_vector_type(8)))  int i32x8;
typedef __attribute__((ext_vector_type(4)))  int i32x4;

#define N_BATCH 1024
#define T_SEQ   60
#define DFEAT   158
#define DMODEL  512
#define NROWS   (N_BATCH * T_SEQ)   // 61440
#define DG      63
#define DG2     126
#define SBATCH  120                  // T_SEQ * 2 spatial heads, batch id = t*2+h

#if defined(__has_builtin)
#if __has_builtin(__builtin_amdgcn_tensor_load_to_lds) && __has_builtin(__builtin_amdgcn_s_wait_tensorcnt)
#define USE_TDM 1
#endif
#endif

__device__ __forceinline__ __bf16 f2bf(float f) {
  union { float f; unsigned u; } a; a.f = f;
  unsigned r = a.u + 0x7FFFu + ((a.u >> 16) & 1u);
  union { unsigned short s; __bf16 b; } c; c.s = (unsigned short)(r >> 16);
  return c.b;
}

__device__ __forceinline__ float gelu_f(float x) {
  return 0.5f * x * (1.0f + erff(x * 0.7071067811865475f));
}

#ifdef USE_TDM
// Issue one TDM 2D tile load: global (row-major, row_stride elements of bf16)
// -> LDS (tile packed contiguously, row by row). D# per CDNA5 ISA 8.3/8.4.
__device__ __forceinline__ void tdm_load_tile(const void* gaddr, unsigned lds_byte_addr,
                                              int tile_k, int tile_rows,
                                              long long row_stride_elems) {
  unsigned long long ga = (unsigned long long)gaddr;
  u32x4 g0;
  g0.x = 1u;                                     // count=1, user D#, no gather
  g0.y = lds_byte_addr;                          // D#.lds_addr (bytes)
  g0.z = (unsigned)(ga & 0xFFFFFFFFull);         // global_addr[31:0]
  g0.w = (unsigned)((ga >> 32) & 0x01FFFFFFull)  // global_addr[56:32]
       | (2u << 30);                             // type = 2 ("image")
  unsigned td0 = (unsigned)row_stride_elems;     // tensor_dim0 (elements)
  unsigned td1 = (unsigned)tile_rows;            // tensor_dim1
  unsigned long long st0 = (unsigned long long)row_stride_elems; // dim0 stride
  i32x8 g1;
  g1[0] = (int)(1u << 16);                               // data_size = 2 bytes
  g1[1] = (int)((td0 & 0xFFFFu) << 16);                  // tensor_dim0[15:0]
  g1[2] = (int)((td0 >> 16) | ((td1 & 0xFFFFu) << 16));  // dim0 hi16 | dim1 lo16
  g1[3] = (int)((td1 >> 16) | ((unsigned)tile_k << 16)); // dim1 hi16 | tile_dim0
  g1[4] = (int)(unsigned)tile_rows;                      // tile_dim1 (tile_dim2=0)
  g1[5] = (int)(unsigned)(st0 & 0xFFFFFFFFull);          // dim0_stride[31:0]
  g1[6] = (int)(unsigned)((st0 >> 32) & 0xFFFFull);      // dim0_stride[47:32]
  g1[7] = 0;                                             // dim1_stride hi (unused, 2D)
  i32x4 zero4 = {0, 0, 0, 0};
  i32x8 zero8 = {0, 0, 0, 0, 0, 0, 0, 0};
  __builtin_amdgcn_tensor_load_to_lds(g0, g1, zero4, zero4, zero8, 0);
}
#endif

// ---------------------------------------------------------------------------
// bf16 WMMA GEMM, LDS double-buffered pipeline.
// C[b] = act(A[b] @ Bt[b]^T + bias (+ resid))
// A: [M][Kp] bf16 row-major, Bt: [N][Kp] bf16 row-major (pre-transposed).
// Block = 256 threads (8 waves) -> 128x64 C tile; wave w -> rows [w*16,w*16+16).
// Per k-step (32): A tile 128x32 (8KB) + B tile 64x32 (4KB) double buffered in
// LDS (24KB). With TDM: wave 0 DMAs tile k+1 (tensor_load_to_lds, TENSORcnt)
// while all 8 waves run WMMAs on tile k from LDS; s_wait_tensorcnt + one
// barrier per k-step. Fallback: per-lane global->reg->ds_store staging.
// REQUIRES: M % 128 == 0, N % 64 == 0, Kp % 32 == 0 (true at all call sites).
// flags: 1=GELU, 2=+resid, 4=+bias. Batched via blockIdx.z + element strides.
// ---------------------------------------------------------------------------
__global__ __launch_bounds__(256) void gemm_bf16_wmma(
    const __bf16* __restrict__ A, const __bf16* __restrict__ Bt,
    const float* __restrict__ bias, const float* __restrict__ resid,
    float* __restrict__ C, int M, int N, int Kp, int ldc,
    long long sA, long long sB, long long sC, int flags)
{
  __shared__ __bf16 lA[2][128 * 32];
  __shared__ __bf16 lB[2][64 * 32];
  const int tid  = threadIdx.x;
  const int wave = tid >> 5;
  const int lane = tid & 31;
  const int m0 = blockIdx.y * 128;
  const int n0 = blockIdx.x * 64;
  const long long bz = blockIdx.z;
  A  += bz * sA;
  Bt += bz * sB;
  C  += bz * sC;
  if (flags & 2) resid += bz * sC;

#ifdef USE_TDM
  const __bf16* gAt = A  + (long long)m0 * Kp;   // A tile base (rows m0..m0+127)
  const __bf16* gBt = Bt + (long long)n0 * Kp;   // B tile base (rows n0..n0+63)
  const unsigned lA_addr[2] = { (unsigned)(unsigned long long)&lA[0][0],
                                (unsigned)(unsigned long long)&lA[1][0] };
  const unsigned lB_addr[2] = { (unsigned)(unsigned long long)&lB[0][0],
                                (unsigned)(unsigned long long)&lB[1][0] };
  if (wave == 0) {                               // stage tile k0=0 via TDM
    tdm_load_tile(gAt, lA_addr[0], 32, 128, Kp);
    tdm_load_tile(gBt, lB_addr[0], 32, 64, Kp);
    __builtin_amdgcn_s_wait_tensorcnt(0);
  }
  __syncthreads();
#else
  // per-thread staging assignment: A 32B, B 16B
  const int ar = tid >> 1, ah = (tid & 1) << 4;
  const int br = tid >> 2, bh = (tid & 3) << 3;
  const __bf16* gA = A  + (long long)(m0 + ar) * Kp + ah;
  const __bf16* gB = Bt + (long long)(n0 + br) * Kp + bh;
  const int lAo = ar * 32 + ah;
  const int lBo = br * 32 + bh;
  uint4 ra0 = *(const uint4*)(gA);
  uint4 ra1 = *(const uint4*)(gA + 8);
  uint4 rb0 = *(const uint4*)(gB);
  *(uint4*)&lA[0][lAo]     = ra0;
  *(uint4*)&lA[0][lAo + 8] = ra1;
  *(uint4*)&lB[0][lBo]     = rb0;
  __syncthreads();
#endif

  const int rc = lane & 15;          // A row in tile / B column in tile
  const int kh = (lane >> 4) << 4;   // K half: 0 or 16
  v8f acc0 = {}, acc1 = {}, acc2 = {}, acc3 = {};
  int cur = 0;
  for (int k0 = 0; k0 < Kp; k0 += 32) {
    const int kn = k0 + 32;
    const bool has_next = kn < Kp;
    if (has_next) {                  // stage tile k+1 while computing tile k
#ifdef USE_TDM
      if (wave == 0) {
        tdm_load_tile(gAt + kn, lA_addr[cur ^ 1], 32, 128, Kp);
        tdm_load_tile(gBt + kn, lB_addr[cur ^ 1], 32, 64, Kp);
      }
#else
      ra0 = *(const uint4*)(gA + kn);
      ra1 = *(const uint4*)(gA + kn + 8);
      rb0 = *(const uint4*)(gB + kn);
      if (kn + 32 < Kp) {
        __builtin_prefetch(gA + kn + 32, 0, 3);
        __builtin_prefetch(gB + kn + 32, 0, 3);
      }
#endif
    }
    // compute tile k0 from LDS (waits only on DScnt)
    const __bf16* la = &lA[cur][((wave << 4) + rc) * 32 + kh];
    const __bf16* lb = &lB[cur][rc * 32 + kh];
    v16bf a  = *(const v16bf*)la;
    v16bf b0 = *(const v16bf*)(lb);
    v16bf b1 = *(const v16bf*)(lb + 16 * 32);
    v16bf b2 = *(const v16bf*)(lb + 32 * 32);
    v16bf b3 = *(const v16bf*)(lb + 48 * 32);
    acc0 = __builtin_amdgcn_wmma_f32_16x16x32_bf16(false, a, false, b0, (short)0, acc0, false, false);
    acc1 = __builtin_amdgcn_wmma_f32_16x16x32_bf16(false, a, false, b1, (short)0, acc1, false, false);
    acc2 = __builtin_amdgcn_wmma_f32_16x16x32_bf16(false, a, false, b2, (short)0, acc2, false, false);
    acc3 = __builtin_amdgcn_wmma_f32_16x16x32_bf16(false, a, false, b3, (short)0, acc3, false, false);
    if (has_next) {
#ifdef USE_TDM
      if (wave == 0) __builtin_amdgcn_s_wait_tensorcnt(0);
      __syncthreads();
#else
      *(uint4*)&lA[cur ^ 1][lAo]     = ra0;
      *(uint4*)&lA[cur ^ 1][lAo + 8] = ra1;
      *(uint4*)&lB[cur ^ 1][lBo]     = rb0;
      __syncthreads();
#endif
      cur ^= 1;
    }
  }

  // C layout: VGPR r, lanes 0-15 -> M=base+r, lanes 16-31 -> M=base+8+r
  const int mb = m0 + (wave << 4) + ((lane >> 4) << 3);
  v8f accs[4] = {acc0, acc1, acc2, acc3};
#pragma unroll
  for (int j = 0; j < 4; ++j) {
    int n = n0 + j * 16 + rc;
    if (n >= N) continue;
    float bv = (flags & 4) ? bias[n] : 0.f;
#pragma unroll
    for (int r = 0; r < 8; ++r) {
      int m = mb + r;
      float vv = accs[j][r] + bv;
      if (flags & 2) vv += resid[(long long)m * ldc + n];
      if (flags & 1) vv = gelu_f(vv);
      C[(long long)m * ldc + n] = vv;
    }
  }
}

// --------------------------- elementwise / pack ----------------------------
__global__ void wtrans_bf16(const float* __restrict__ w, __bf16* __restrict__ wt,
                            int K, int N, int Kp) {
  long long idx = (long long)blockIdx.x * blockDim.x + threadIdx.x;
  long long total = (long long)N * Kp;
  if (idx >= total) return;
  long long n = idx / Kp;
  int kk = (int)(idx - n * Kp);
  wt[idx] = (kk < K) ? f2bf(w[(long long)kk * N + n]) : f2bf(0.f);
}

__global__ void pack_bf16(const float* __restrict__ src, __bf16* __restrict__ dst,
                          long long rows, int K, int Kp) {
  long long idx = (long long)blockIdx.x * blockDim.x + threadIdx.x;
  long long total = rows * Kp;
  if (idx >= total) return;
  long long r = idx / Kp;
  int c = (int)(idx - r * Kp);
  dst[idx] = (c < K) ? f2bf(src[r * K + c]) : f2bf(0.f);
}

__global__ void gate_apply_pack(const float* __restrict__ x, const float* __restrict__ gate,
                                __bf16* __restrict__ dst) {
  long long idx = (long long)blockIdx.x * blockDim.x + threadIdx.x;
  const long long total = (long long)NROWS * 160;   // K 158 padded to 160
  if (idx >= total) return;
  long long row = idx / 160;
  int c = (int)(idx - row * 160);
  float vv = 0.f;
  if (c < DFEAT) {
    long long n = row / T_SEQ;
    vv = x[row * 221 + c] * gate[n * DFEAT + c];
  }
  dst[idx] = f2bf(vv);
}

__global__ void add_pos(float* __restrict__ z) {
  long long idx = (long long)blockIdx.x * blockDim.x + threadIdx.x;
  if (idx >= (long long)NROWS * DMODEL) return;
  int d = (int)(idx & 511);
  long long row = idx >> 9;
  int t = (int)(row % T_SEQ);
  float div = expf((float)((d >> 1) * 2) * (-9.210340371976184f / 512.0f));
  float arg = (float)t * div;
  z[idx] += (d & 1) ? cosf(arg) : sinf(arg);
}

// ----------------------------- LayerNorm (wave/row) ------------------------
__global__ __launch_bounds__(256) void ln_kernel(
    const float* __restrict__ x, const float* __restrict__ res,
    const float* __restrict__ g, const float* __restrict__ b,
    float* __restrict__ y, long long rows, int D)
{
  int wave = threadIdx.x >> 5, lane = threadIdx.x & 31;
  long long row = (long long)blockIdx.x * 8 + wave;
  if (row >= rows) return;
  const float* xr = x + row * D;
  const float* rr = res ? res + row * D : nullptr;
  float vals[16];                 // D == 512 -> 16 per lane
  int nper = D >> 5;
  float s = 0.f;
  for (int i = 0; i < nper; ++i) {
    float v = xr[lane + (i << 5)];
    if (rr) v += rr[lane + (i << 5)];
    vals[i] = v; s += v;
  }
  for (int o = 16; o >= 1; o >>= 1) s += __shfl_xor(s, o, 32);
  float mean = s / (float)D;
  float vs = 0.f;
  for (int i = 0; i < nper; ++i) { float d = vals[i] - mean; vs += d * d; }
  for (int o = 16; o >= 1; o >>= 1) vs += __shfl_xor(vs, o, 32);
  float inv = rsqrtf(vs / (float)D + 1e-5f);
  float* yr = y + row * D;
  for (int i = 0; i < nper; ++i) {
    int c = lane + (i << 5);
    yr[c] = (vals[i] - mean) * inv * g[c] + b[c];
  }
}

// ----------------------------- gating MLP ----------------------------------
__global__ __launch_bounds__(64) void gate_kernel(
    const float* __restrict__ x, const float* __restrict__ w1, const float* __restrict__ b1,
    const float* __restrict__ w2, const float* __restrict__ b2, float* __restrict__ gate)
{
  __shared__ float gin[DG];
  __shared__ float h1[DG2];
  __shared__ float h[DFEAT];
  __shared__ float temp_s, mx_s, sum_s;
  int n = blockIdx.x, tid = threadIdx.x;
  const float* xi = x + ((long long)n * T_SEQ + (T_SEQ - 1)) * 221 + DFEAT;
  for (int i = tid; i < DG; i += 64) gin[i] = xi[i];
  __syncthreads();
  if (tid == 0) {
    float m = 0.f;
    for (int i = 0; i < DG; ++i) m += gin[i];
    m /= (float)DG;
    float v = 0.f;
    for (int i = 0; i < DG; ++i) { float d = gin[i] - m; v += d * d; }
    float vol = sqrtf(v / (float)(DG - 1));
    temp_s = 2.0f * (1.0f + 1.0f / (1.0f + expf(-vol)));
  }
  for (int j = tid; j < DG2; j += 64) {
    float s = b1[j];
    for (int i = 0; i < DG; ++i) s += gin[i] * w1[i * DG2 + j];
    h1[j] = gelu_f(s);
  }
  __syncthreads();
  for (int j = tid; j < DFEAT; j += 64) {
    float s = b2[j];
    for (int i = 0; i < DG2; ++i) s += h1[i] * w2[i * DFEAT + j];
    h[j] = s;
  }
  __syncthreads();
  if (tid == 0) {
    float t = temp_s, mx = -1e30f;
    for (int i = 0; i < DFEAT; ++i) mx = fmaxf(mx, h[i] / t);
    float s = 0.f;
    for (int i = 0; i < DFEAT; ++i) s += expf(h[i] / t - mx);
    mx_s = mx; sum_s = s;
  }
  __syncthreads();
  float t = temp_s;
  for (int j = tid; j < DFEAT; j += 64)
    gate[(long long)n * DFEAT + j] = 158.0f * expf(h[j] / t - mx_s) / sum_s;
}

// ------------------------ temporal causal attention ------------------------
__global__ __launch_bounds__(64) void temporal_attn(
    const float* __restrict__ q, const float* __restrict__ k,
    const float* __restrict__ v, float* __restrict__ att)
{
  __shared__ float ks[T_SEQ * 128];
  __shared__ float ss[T_SEQ * T_SEQ];
  int n = blockIdx.x, h = blockIdx.y;
  const long long rowbase = (long long)n * T_SEQ;
  const int hoff = h * 128;
  for (int i = threadIdx.x; i < T_SEQ * 128; i += 64) {
    int t = i >> 7, d = i & 127;
    ks[i] = k[(rowbase + t) * DMODEL + hoff + d];
  }
  __syncthreads();
  int t = threadIdx.x;
  if (t < T_SEQ) {
    const float* qr = q + (rowbase + t) * DMODEL + hoff;
    for (int s = 0; s <= t; ++s) {
      float acc = 0.f;
      const float* kr = &ks[s * 128];
#pragma unroll 8
      for (int d = 0; d < 128; ++d) acc += qr[d] * kr[d];
      ss[t * T_SEQ + s] = acc;
    }
    float mx = -1e30f;
    for (int s = 0; s <= t; ++s) mx = fmaxf(mx, ss[t * T_SEQ + s]);
    float sum = 0.f;
    for (int s = 0; s <= t; ++s) { float e = expf(ss[t * T_SEQ + s] - mx); ss[t * T_SEQ + s] = e; sum += e; }
    float inv = 1.f / sum;
    for (int s = 0; s <= t; ++s) ss[t * T_SEQ + s] *= inv;
  }
  __syncthreads();
  if (t < T_SEQ) {
    float* ar = att + (rowbase + t) * DMODEL + hoff;
    for (int d = 0; d < 128; ++d) {
      float acc = 0.f;
      for (int s = 0; s <= t; ++s) acc += ss[t * T_SEQ + s] * v[(rowbase + s) * DMODEL + hoff + d];
      ar[d] = acc;
    }
  }
}

// ----------------------- spatial attention helpers -------------------------
__global__ void pack_qk_spatial(const float* __restrict__ q, __bf16* __restrict__ qs) {
  // qs[(t*2+h)][n][d], d in 0..255
  long long idx = (long long)blockIdx.x * blockDim.x + threadIdx.x;
  if (idx >= (long long)SBATCH * 1024 * 256) return;
  long long b = idx >> 18;
  int rem = (int)(idx & 262143);
  int n = rem >> 8, d = rem & 255;
  int t = (int)(b >> 1), h = (int)(b & 1);
  qs[idx] = f2bf(q[((long long)n * T_SEQ + t) * DMODEL + h * 256 + d]);
}

__global__ void pack_v_spatial(const float* __restrict__ v, __bf16* __restrict__ vt) {
  // vt[(t*2+h)][d][m]  (transposed so W@V GEMM's B-fragments are contiguous)
  long long idx = (long long)blockIdx.x * blockDim.x + threadIdx.x;
  if (idx >= (long long)SBATCH * 256 * 1024) return;
  long long b = idx >> 18;
  int rem = (int)(idx & 262143);
  int d = rem >> 10, m = rem & 1023;
  int t = (int)(b >> 1), h = (int)(b & 1);
  vt[idx] = f2bf(v[((long long)m * T_SEQ + t) * DMODEL + h * 256 + d]);
}

__global__ void zero_stats(float* __restrict__ stats) {
  if (threadIdx.x < 4) stats[threadIdx.x] = 0.f;
}

__global__ void score_stats(const float* __restrict__ s, float* __restrict__ stats, long long total) {
  long long i = (long long)blockIdx.x * blockDim.x + threadIdx.x;
  long long stride = (long long)gridDim.x * blockDim.x;
  float s0 = 0, q0 = 0, s1 = 0, q1 = 0;
  for (; i < total; i += stride) {
    float vv = s[i];
    if (((i >> 20) & 1) == 0) { s0 += vv; q0 += vv * vv; }
    else                      { s1 += vv; q1 += vv * vv; }
  }
  for (int o = 16; o >= 1; o >>= 1) {
    s0 += __shfl_xor(s0, o, 32); q0 += __shfl_xor(q0, o, 32);
    s1 += __shfl_xor(s1, o, 32); q1 += __shfl_xor(q1, o, 32);
  }
  if ((threadIdx.x & 31) == 0) {
    atomicAdd(&stats[0], s0); atomicAdd(&stats[1], q0);
    atomicAdd(&stats[2], s1); atomicAdd(&stats[3], q1);
  }
}

__global__ __launch_bounds__(256) void spatial_softmax(
    const float* __restrict__ s, const float* __restrict__ stats, __bf16* __restrict__ w)
{
  int wave = threadIdx.x >> 5, lane = threadIdx.x & 31;
  long long row = (long long)blockIdx.x * 8 + wave;
  if (row >= (long long)SBATCH * 1024) return;
  int h = (int)((row >> 10) & 1);
  const float cnt = 60.0f * 1024.0f * 1024.0f;
  float sm = stats[h * 2], sq = stats[h * 2 + 1];
  float var = (sq - sm * sm / cnt) / (cnt - 1.0f);
  float inv = 1.0f / (16.0f * sqrtf(1.0f + sqrtf(fmaxf(var, 0.f))));
  const float* sr = s + (row << 10);
  float vals[32];
  float mx = -1e30f;
#pragma unroll
  for (int i = 0; i < 32; ++i) { float vv = sr[lane + i * 32] * inv; vals[i] = vv; mx = fmaxf(mx, vv); }
  for (int o = 16; o >= 1; o >>= 1) mx = fmaxf(mx, __shfl_xor(mx, o, 32));
  float sum = 0.f;
#pragma unroll
  for (int i = 0; i < 32; ++i) { float e = expf(vals[i] - mx); vals[i] = e; sum += e; }
  for (int o = 16; o >= 1; o >>= 1) sum += __shfl_xor(sum, o, 32);
  float invs = 1.0f / sum;
  __bf16* wr = w + (row << 10);
#pragma unroll
  for (int i = 0; i < 32; ++i) wr[lane + i * 32] = f2bf(vals[i] * invs);
}

// ------------------------------ pooling / head -----------------------------
__global__ __launch_bounds__(256) void base_dot(const float* __restrict__ ht, float* __restrict__ base) {
  int wave = threadIdx.x >> 5, lane = threadIdx.x & 31;
  long long row = (long long)blockIdx.x * 8 + wave;
  if (row >= NROWS) return;
  int t = (int)(row % T_SEQ);
  long long n = row / T_SEQ;
  const float* hr = ht + row * DMODEL;
  const float* qr = ht + (n * T_SEQ + (T_SEQ - 1)) * DMODEL;
  float s = 0.f;
  for (int i = lane; i < DMODEL; i += 32) s += hr[i] * qr[i];
  for (int o = 16; o >= 1; o >>= 1) s += __shfl_xor(s, o, 32);
  if (lane == 0) base[row] = s * expf(-0.1f * (float)(T_SEQ - 1 - t));
}

__global__ void pool_softmax(const float* __restrict__ base, float* __restrict__ w) {
  int n = blockIdx.x;
  if (threadIdx.x == 0) {
    float mx = -1e30f;
    for (int t = 0; t < T_SEQ; ++t) mx = fmaxf(mx, base[n * T_SEQ + t]);
    float s = 0.f;
    for (int t = 0; t < T_SEQ; ++t) { float e = expf(base[n * T_SEQ + t] - mx); w[n * T_SEQ + t] = e; s += e; }
    float inv = 1.f / s;
    for (int t = 0; t < T_SEQ; ++t) w[n * T_SEQ + t] *= inv;
  }
}

__global__ void pool_apply(const float* __restrict__ w, const float* __restrict__ z, float* __restrict__ pooled) {
  long long idx = (long long)blockIdx.x * blockDim.x + threadIdx.x;
  if (idx >= (long long)N_BATCH * DMODEL) return;
  long long n = idx >> 9;
  int d = (int)(idx & 511);
  const float* zr = z + (n * T_SEQ) * DMODEL + d;
  const float* wr = w + n * T_SEQ;
  float acc = 0.f;
  for (int t = 0; t < T_SEQ; ++t) acc += wr[t] * zr[(long long)t * DMODEL];
  pooled[idx] = acc;
}

__global__ __launch_bounds__(256) void head2_kernel(
    const float* __restrict__ hb, const float* __restrict__ w2,
    const float* __restrict__ b2, float* __restrict__ out)
{
  int wave = threadIdx.x >> 5, lane = threadIdx.x & 31;
  int n = blockIdx.x * 8 + wave;
  if (n >= N_BATCH) return;
  float s = 0.f;
  for (int i = lane; i < 256; i += 32) s += hb[(long long)n * 256 + i] * w2[i];
  for (int o = 16; o >= 1; o >>= 1) s += __shfl_xor(s, o, 32);
  if (lane == 0) out[n] = s + b2[0];
}

// ===========================================================================
extern "C" void kernel_launch(void* const* d_in, const int* in_sizes, int n_in,
                              void* d_out, int out_size, void* d_ws, size_t ws_size,
                              hipStream_t stream) {
  (void)in_sizes; (void)n_in; (void)out_size; (void)ws_size;
  enum {
    IN_X = 0, IN_GW1, IN_GB1, IN_GW2, IN_GB2, IN_INW, IN_INB,
    IN_TEMPW, IN_TEMPB, IN_H1W, IN_H1B, IN_H2W, IN_H2B,
    IN_TQW, IN_TQB, IN_TKW, IN_TKB, IN_TVW, IN_TVB,
    IN_TN1G, IN_TN1B, IN_TN2G, IN_TN2B, IN_TF1W, IN_TF1B, IN_TF2W, IN_TF2B,
    IN_SQW, IN_SQB, IN_SKW, IN_SKB, IN_SVW, IN_SVB,
    IN_SN1G, IN_SN1B, IN_SN2G, IN_SN2B, IN_SF1W, IN_SF1B, IN_SF2W, IN_SF2B
  };
#define F(i) ((const float*)d_in[i])

  // ---- workspace layout (bump allocator, 256B aligned) ----
  char* wsb = (char*)d_ws;
  size_t off = 0;
  auto alloc = [&](size_t bytes) -> char* {
    off = (off + 255) & ~(size_t)255;
    char* p = wsb + off; off += bytes; return p;
  };
  float*  gate   = (float*) alloc((size_t)N_BATCH * DFEAT * 4);
  float*  z      = (float*) alloc((size_t)NROWS * DMODEL * 4);
  float*  xn     = (float*) alloc((size_t)NROWS * DMODEL * 4);
  float*  attb   = (float*) alloc((size_t)NROWS * DMODEL * 4);
  float*  xt     = (float*) alloc((size_t)NROWS * DMODEL * 4);
  float*  qb     = (float*) alloc((size_t)NROWS * DMODEL * 4);
  float*  kb     = (float*) alloc((size_t)NROWS * DMODEL * 4);
  float*  vb     = (float*) alloc((size_t)NROWS * DMODEL * 4);
  float*  hbuf1  = (float*) alloc((size_t)NROWS * 1024 * 4);
  __bf16* Abf    = (__bf16*)alloc((size_t)NROWS * 1024 * 2);
  __bf16* Qs     = (__bf16*)alloc((size_t)SBATCH * 1024 * 256 * 2);
  __bf16* Ks     = (__bf16*)alloc((size_t)SBATCH * 1024 * 256 * 2);
  __bf16* Vt     = (__bf16*)alloc((size_t)SBATCH * 256 * 1024 * 2);
  float*  Ssc    = (float*) alloc((size_t)SBATCH * 1024 * 1024 * 4);
  __bf16* Wbf    = (__bf16*)alloc((size_t)SBATCH * 1024 * 1024 * 2);
  float*  stats  = (float*) alloc(16);
  float*  basep  = (float*) alloc((size_t)NROWS * 4);
  float*  wsoft  = (float*) alloc((size_t)NROWS * 4);
  float*  pooled = (float*) alloc((size_t)N_BATCH * DMODEL * 4);
  float*  hsmall = (float*) alloc((size_t)N_BATCH * 256 * 4);
  __bf16* w_in_t   = (__bf16*)alloc((size_t)512 * 160 * 2);
  __bf16* w_temp_t = (__bf16*)alloc((size_t)512 * 512 * 2);
  __bf16* w_h1_t   = (__bf16*)alloc((size_t)256 * 512 * 2);
  __bf16* w_qt[2], *w_kt[2], *w_vt[2], *w_f1t[2], *w_f2t[2];
  for (int b = 0; b < 2; ++b) {
    w_qt[b]  = (__bf16*)alloc((size_t)512 * 512 * 2);
    w_kt[b]  = (__bf16*)alloc((size_t)512 * 512 * 2);
    w_vt[b]  = (__bf16*)alloc((size_t)512 * 512 * 2);
    w_f1t[b] = (__bf16*)alloc((size_t)1024 * 512 * 2);
    w_f2t[b] = (__bf16*)alloc((size_t)512 * 1024 * 2);
  }

  auto g1 = [](long long n) { return (unsigned)((n + 255) / 256); };
  auto wt = [&](const float* w, __bf16* dst, int K, int N, int Kp) {
    wtrans_bf16<<<g1((long long)N * Kp), 256, 0, stream>>>(w, dst, K, N, Kp);
  };
  auto pack = [&](const float* src, __bf16* dst, long long rows, int K, int Kp) {
    pack_bf16<<<g1(rows * Kp), 256, 0, stream>>>(src, dst, rows, K, Kp);
  };
  auto gemm = [&](const __bf16* Ab, const __bf16* Bt, const float* bias, const float* resid,
                  float* Cc, int M, int Nn, int Kp, int ldc,
                  long long sA, long long sB, long long sC, int nb, int flags) {
    dim3 g(Nn / 64, (M + 127) / 128, nb);
    gemm_bf16_wmma<<<g, 256, 0, stream>>>(Ab, Bt, bias, resid, Cc, M, Nn, Kp, ldc, sA, sB, sC, flags);
  };

  // ---- weight transposes (f32 [K][N] -> bf16 [N][Kpad]) ----
  wt(F(IN_INW),   w_in_t,   158,  512, 160);
  wt(F(IN_TEMPW), w_temp_t, 512,  512, 512);
  wt(F(IN_H1W),   w_h1_t,   512,  256, 512);
  wt(F(IN_TQW),  w_qt[0], 512, 512, 512);  wt(F(IN_TKW), w_kt[0], 512, 512, 512);
  wt(F(IN_TVW),  w_vt[0], 512, 512, 512);
  wt(F(IN_TF1W), w_f1t[0], 512, 1024, 512); wt(F(IN_TF2W), w_f2t[0], 1024, 512, 1024);
  wt(F(IN_SQW),  w_qt[1], 512, 512, 512);  wt(F(IN_SKW), w_kt[1], 512, 512, 512);
  wt(F(IN_SVW),  w_vt[1], 512, 512, 512);
  wt(F(IN_SF1W), w_f1t[1], 512, 1024, 512); wt(F(IN_SF2W), w_f2t[1], 1024, 512, 1024);

  // ---- gating + input projection + positional encoding ----
  gate_kernel<<<N_BATCH, 64, 0, stream>>>(F(IN_X), F(IN_GW1), F(IN_GB1), F(IN_GW2), F(IN_GB2), gate);
  gate_apply_pack<<<g1((long long)NROWS * 160), 256, 0, stream>>>(F(IN_X), gate, Abf);
  gemm(Abf, w_in_t, F(IN_INB), nullptr, z, NROWS, 512, 160, 512, 0, 0, 0, 1, 4);
  add_pos<<<g1((long long)NROWS * DMODEL), 256, 0, stream>>>(z);

  const unsigned lnb = (NROWS + 7) / 8;

  // ---- temporal block ----
  ln_kernel<<<lnb, 256, 0, stream>>>(z, nullptr, F(IN_TN1G), F(IN_TN1B), xn, NROWS, DMODEL);
  pack(xn, Abf, NROWS, 512, 512);
  gemm(Abf, w_qt[0], F(IN_TQB), nullptr, qb, NROWS, 512, 512, 512, 0, 0, 0, 1, 4);
  gemm(Abf, w_kt[0], F(IN_TKB), nullptr, kb, NROWS, 512, 512, 512, 0, 0, 0, 1, 4);
  gemm(Abf, w_vt[0], F(IN_TVB), nullptr, vb, NROWS, 512, 512, 512, 0, 0, 0, 1, 4);
  temporal_attn<<<dim3(N_BATCH, 4), 64, 0, stream>>>(qb, kb, vb, attb);
  ln_kernel<<<lnb, 256, 0, stream>>>(xn, attb, F(IN_TN2G), F(IN_TN2B), xt, NROWS, DMODEL);
  pack(xt, Abf, NROWS, 512, 512);
  gemm(Abf, w_f1t[0], F(IN_TF1B), nullptr, hbuf1, NROWS, 1024, 512, 1024, 0, 0, 0, 1, 4 | 1);
  pack(hbuf1, Abf, NROWS, 1024, 1024);
  gemm(Abf, w_f2t[0], F(IN_TF2B), xt, z, NROWS, 512, 1024, 512, 0, 0, 0, 1, 4 | 2);

  // ---- spatial block ----
  ln_kernel<<<lnb, 256, 0, stream>>>(z, nullptr, F(IN_SN1G), F(IN_SN1B), xn, NROWS, DMODEL);
  pack(xn, Abf, NROWS, 512, 512);
  gemm(Abf, w_qt[1], F(IN_SQB), nullptr, qb, NROWS, 512, 512, 512, 0, 0, 0, 1, 4);
  gemm(Abf, w_kt[1], F(IN_SKB), nullptr, kb, NROWS, 512, 512, 512, 0, 0, 0, 1, 4);
  gemm(Abf, w_vt[1], F(IN_SVB), nullptr, vb, NROWS, 512, 512, 512, 0, 0, 0, 1, 4);
  pack_qk_spatial<<<g1((long long)SBATCH * 1024 * 256), 256, 0, stream>>>(qb, Qs);
  pack_qk_spatial<<<g1((long long)SBATCH * 1024 * 256), 256, 0, stream>>>(kb, Ks);
  // scores[(t,h)][n][m] = Q . K  (120 batched 1024x1024x256 WMMA GEMMs)
  gemm(Qs, Ks, nullptr, nullptr, Ssc, 1024, 1024, 256, 1024,
       1024LL * 256, 1024LL * 256, 1LL << 20, SBATCH, 0);
  zero_stats<<<1, 32, 0, stream>>>(stats);
  score_stats<<<4096, 256, 0, stream>>>(Ssc, stats, (long long)SBATCH << 20);
  spatial_softmax<<<(SBATCH * 1024) / 8, 256, 0, stream>>>(Ssc, stats, Wbf);
  pack_v_spatial<<<g1((long long)SBATCH * 256 * 1024), 256, 0, stream>>>(vb, Vt);
  // att[n][t*512 + h*256 + d] = W @ V  (batch offset (t*2+h)*256 == strideC*z)
  gemm(Wbf, Vt, nullptr, nullptr, attb, 1024, 256, 1024, T_SEQ * DMODEL,
       1LL << 20, 256LL * 1024, 256, SBATCH, 0);
  ln_kernel<<<lnb, 256, 0, stream>>>(xn, attb, F(IN_SN2G), F(IN_SN2B), xt, NROWS, DMODEL);
  pack(xt, Abf, NROWS, 512, 512);
  gemm(Abf, w_f1t[1], F(IN_SF1B), nullptr, hbuf1, NROWS, 1024, 512, 1024, 0, 0, 0, 1, 4 | 1);
  pack(hbuf1, Abf, NROWS, 1024, 1024);
  gemm(Abf, w_f2t[1], F(IN_SF2B), xt, z, NROWS, 512, 1024, 512, 0, 0, 0, 1, 4 | 2);

  // ---- recency pooling + head ----
  pack(z, Abf, NROWS, 512, 512);
  gemm(Abf, w_temp_t, F(IN_TEMPB), nullptr, qb /* ht */, NROWS, 512, 512, 512, 0, 0, 0, 1, 4);
  base_dot<<<lnb, 256, 0, stream>>>(qb, basep);
  pool_softmax<<<N_BATCH, 32, 0, stream>>>(basep, wsoft);
  pool_apply<<<g1((long long)N_BATCH * DMODEL), 256, 0, stream>>>(wsoft, z, pooled);
  pack(pooled, Abf, N_BATCH, 512, 512);
  gemm(Abf, w_h1_t, F(IN_H1B), nullptr, hsmall, N_BATCH, 256, 512, 256, 0, 0, 0, 1, 4 | 1);
  head2_kernel<<<N_BATCH / 8, 256, 0, stream>>>(hsmall, F(IN_H2W), F(IN_H2B), (float*)d_out);
#undef F
}
